// LSTMCellModel_24807731102263
// MI455X (gfx1250) — matrix operand order repaired and checked by reading, not compiled, not run on hardware
//
#include <hip/hip_runtime.h>

// Fused LSTM cell on gfx1250: z = x@Wx^T + h0@Wh^T + b (4 gates) -> gate math -> h1,c1.
// B=32768, E=H=512. bf16 WMMA 16x16x32 with f32 accumulation; f32->bf16 conversion
// (hw v_cvt_pk_bf16_f32) staged through double-buffered LDS.
// Workgroup = 128 rows x 32 h-cols x 4 gates; 8 waves in a 4(M)x2(N) grid.

#define B_DIM 32768
#define K_DIM 512   // E == H == 512
#define H_DIM 512

typedef __attribute__((ext_vector_type(16))) __bf16 v16bf;
typedef __attribute__((ext_vector_type(8)))  float  v8f;

union Frag {
  v16bf v;
  uint4 q[2];
};

// Two f32 -> packed bf16 pair (RNE); backend selects v_cvt_pk_bf16_f32.
__device__ __forceinline__ unsigned int pk2(float a, float b) {
  union { __bf16 h[2]; unsigned int u; } r;
  r.h[0] = (__bf16)a;
  r.h[1] = (__bf16)b;
  return r.u;
}

__device__ __forceinline__ void cvt16(const float4* f, uint4& lo, uint4& hi) {
  lo.x = pk2(f[0].x, f[0].y); lo.y = pk2(f[0].z, f[0].w);
  lo.z = pk2(f[1].x, f[1].y); lo.w = pk2(f[1].z, f[1].w);
  hi.x = pk2(f[2].x, f[2].y); hi.y = pk2(f[2].z, f[2].w);
  hi.z = pk2(f[3].x, f[3].y); hi.w = pk2(f[3].z, f[3].w);
}

__device__ __forceinline__ float sig_f(float x) {
  return __builtin_amdgcn_rcpf(1.0f + __expf(-x));
}
__device__ __forceinline__ float tanh_f(float x) {
  return 1.0f - 2.0f * __builtin_amdgcn_rcpf(__expf(2.0f * x) + 1.0f);
}

__global__ __launch_bounds__(256, 1)
void lstm_fused_wmma(const float* __restrict__ X,  const float* __restrict__ H0,
                     const float* __restrict__ C0, const float* __restrict__ Wx,
                     const float* __restrict__ bx, const float* __restrict__ Wh,
                     const float* __restrict__ bh, float* __restrict__ H1,
                     float* __restrict__ C1) {
  // Double-buffered tiles (bf16): A 128x32, W 4x32x32. 2*(8KB+8KB)=32KB LDS.
  __shared__ unsigned short As[2][128 * 32];
  __shared__ unsigned short Bs[2][4 * 32 * 32];

  const int t    = threadIdx.x;
  const int rb   = blockIdx.x;        // row block   (128 rows)
  const int hb   = blockIdx.y;        // h-col block (32 cols)
  const int lane = t & 31;
  const int wave = t >> 5;
  const int mw   = wave & 3;          // M wave: rows mw*32
  const int nw   = wave >> 2;         // N wave: cols nw*16
  const int lcol = lane & 15;
  const int hiH  = lane >> 4;

  v8f acc[4][2];
  {
    v8f z = {};
    #pragma unroll
    for (int g = 0; g < 4; ++g)
      #pragma unroll
      for (int m = 0; m < 2; ++m)
        acc[g][m] = z;
  }

  // Staging split: A tile 4096 f32 / 256 thr = 16 each; same for W tile.
  const int ar = t >> 1;              // 0..127 A row
  const int ah = (t & 1) << 4;        // 0/16 k-offset
  const int wg = t >> 6;              // gate 0..3
  const int wc = (t >> 1) & 31;       // W col 0..31

  // Per-thread base offsets; carried pointers advance 32 floats per K-step,
  // with a single uniform swap to the H0/Wh phase at ks==16.
  const size_t aoff = (size_t)(rb * 128 + ar) * K_DIM + ah;
  const size_t woff = (size_t)wg * H_DIM * K_DIM + (size_t)(hb * 32 + wc) * K_DIM + ah;
  const float* pa = X  + aoff;
  const float* pw = Wx + woff;

  // LDS staging addresses (constant across the loop)
  const int asA = ar * 32 + ah;
  const int asB = (wg * 32 + wc) * 32 + ah;

  // ---- prologue: global-load tile 0, convert, store to buffer 0 ----
  float4 fa[4], fw[4];
  {
    const float4* qa = (const float4*)pa;
    fa[0] = qa[0]; fa[1] = qa[1]; fa[2] = qa[2]; fa[3] = qa[3];
    const float4* qw = (const float4*)pw;
    fw[0] = qw[0]; fw[1] = qw[1]; fw[2] = qw[2]; fw[3] = qw[3];
    uint4 lo, hi;
    cvt16(fa, lo, hi);
    *(uint4*)&As[0][asA]     = lo;
    *(uint4*)&As[0][asA + 8] = hi;
    cvt16(fw, lo, hi);
    *(uint4*)&Bs[0][asB]     = lo;
    *(uint4*)&Bs[0][asB + 8] = hi;
  }
  // warm epilogue lines while the K-loop runs (global_prefetch_b8, no counters)
  {
    const int h  = hb * 32 + nw * 16 + lcol;
    const int r0 = rb * 128 + mw * 32 + (hiH << 3);
    __builtin_prefetch(&bx[h], 0, 3);
    __builtin_prefetch(&bh[h], 0, 3);
    __builtin_prefetch(&C0[(size_t)r0 * H_DIM + h], 0, 0);
    __builtin_prefetch(&C0[(size_t)(r0 + 16) * H_DIM + h], 0, 0);
  }
  __syncthreads();

  #pragma unroll 2
  for (int ks = 0; ks < 32; ++ks) {
    const int  cur  = ks & 1;
    const bool more = (ks + 1) < 32;

    // ---- issue next tile's global loads early (latency hidden by WMMA) ----
    if (more) {
      if (ks + 1 == 16) {            // uniform phase switch: x/Wx -> h0/Wh
        pa = H0 + aoff;
        pw = Wh + woff;
      } else {
        pa += 32;
        pw += 32;
      }
      const float4* qa = (const float4*)pa;
      fa[0] = qa[0]; fa[1] = qa[1]; fa[2] = qa[2]; fa[3] = qa[3];
      const float4* qw = (const float4*)pw;
      fw[0] = qw[0]; fw[1] = qw[1]; fw[2] = qw[2]; fw[3] = qw[3];
    }

    // ---- compute from buffer `cur` ----
    // A 16x32 layout: low lanes M=lane, K[0:8)+K[16:24); high lanes K[8:16)+K[24:32)
    const int kA = hiH << 3;
    // B 32x16 layout: lane col = lane%16; low lanes K[0:8)+K[16:24); high lanes K[8:16)+K[24:32)
    const int kB = hiH << 3;

    Frag a0, a1;
    const int arow0 = (mw * 32 + lcol) * 32;
    a0.q[0] = *(const uint4*)&As[cur][arow0 + kA];
    a0.q[1] = *(const uint4*)&As[cur][arow0 + kA + 16];
    const int arow1 = arow0 + 16 * 32;
    a1.q[0] = *(const uint4*)&As[cur][arow1 + kA];
    a1.q[1] = *(const uint4*)&As[cur][arow1 + kA + 16];

    #pragma unroll
    for (int g = 0; g < 4; ++g) {
      Frag bf;
      const int brow = (g * 32 + nw * 16 + lcol) * 32;
      bf.q[0] = *(const uint4*)&Bs[cur][brow + kB];
      bf.q[1] = *(const uint4*)&Bs[cur][brow + kB + 16];
      acc[g][0] = __builtin_amdgcn_wmma_f32_16x16x32_bf16(
          false, a0.v, false, bf.v, (short)0, acc[g][0], false, false);
      acc[g][1] = __builtin_amdgcn_wmma_f32_16x16x32_bf16(
          false, a1.v, false, bf.v, (short)0, acc[g][1], false, false);
    }

    // ---- convert + store next tile into the other buffer ----
    if (more) {
      const int nxt = cur ^ 1;
      uint4 lo, hi;
      cvt16(fa, lo, hi);
      *(uint4*)&As[nxt][asA]     = lo;
      *(uint4*)&As[nxt][asA + 8] = hi;
      cvt16(fw, lo, hi);
      *(uint4*)&Bs[nxt][asB]     = lo;
      *(uint4*)&Bs[nxt][asB + 8] = hi;
    }
    __syncthreads();
  }

  // ---- epilogue: bias + gate math + store ----
  // C/D layout: lane -> N = lane%16 ; VGPR j -> M = j + 8*(lane>=16)
  const int h = hb * 32 + nw * 16 + lcol;
  float bias[4];
  #pragma unroll
  for (int g = 0; g < 4; ++g)
    bias[g] = bx[g * H_DIM + h] + bh[g * H_DIM + h];

  #pragma unroll
  for (int m = 0; m < 2; ++m) {
    const int rbase = rb * 128 + mw * 32 + m * 16 + (hiH << 3);
    #pragma unroll
    for (int j = 0; j < 8; ++j) {
      const size_t idx = (size_t)(rbase + j) * H_DIM + h;
      const float zi = acc[0][m][j] + bias[0];
      const float zf = acc[1][m][j] + bias[1];
      const float zo = acc[2][m][j] + bias[2];
      const float zg = acc[3][m][j] + bias[3];
      const float iv = sig_f(zi);
      const float fv = sig_f(zf);
      const float ov = sig_f(zo);
      const float gv = tanh_f(zg);
      const float c0v = __builtin_nontemporal_load(&C0[idx]);
      const float c1  = fv * c0v + iv * gv;
      __builtin_nontemporal_store(c1, &C1[idx]);
      __builtin_nontemporal_store(ov * tanh_f(c1), &H1[idx]);
    }
  }
}

extern "C" void kernel_launch(void* const* d_in, const int* in_sizes, int n_in,
                              void* d_out, int out_size, void* d_ws, size_t ws_size,
                              hipStream_t stream) {
  (void)in_sizes; (void)n_in; (void)out_size; (void)d_ws; (void)ws_size;
  const float* X  = (const float*)d_in[0];
  const float* H0 = (const float*)d_in[1];
  const float* C0 = (const float*)d_in[2];
  const float* Wx = (const float*)d_in[3];
  const float* bx = (const float*)d_in[4];
  const float* Wh = (const float*)d_in[5];
  const float* bh = (const float*)d_in[6];
  float* H1 = (float*)d_out;                              // [B,H]
  float* C1 = (float*)d_out + (size_t)B_DIM * H_DIM;      // [B,H]

  dim3 grid(B_DIM / 128, H_DIM / 32);   // 256 x 16 workgroups
  lstm_fused_wmma<<<grid, 256, 0, stream>>>(X, H0, C0, Wx, bx, Wh, bh, H1, C1);
}
